// BidirectionalRNN_24206435680653
// MI455X (gfx1250) — matrix operand order repaired and checked by reading, not compiled, not run on hardware
//
#include <hip/hip_runtime.h>
#include <hip/hip_bf16.h>
#include <stdint.h>

#define T_SEQ 32768
#define HD 512

typedef __bf16 bf16x16 __attribute__((ext_vector_type(16)));
typedef float  f32x8   __attribute__((ext_vector_type(8)));

union Frag16 { uint4 q[2]; uint32_t u[8]; bf16x16 v; };
union CFrag  { f32x8 v; float f[8]; };

__device__ __forceinline__ uint32_t f2bf(float f) {
    uint32_t u = __float_as_uint(f);
    return (u + 0x7FFFu + ((u >> 16) & 1u)) >> 16;   // round-to-nearest-even
}
__device__ __forceinline__ uint32_t pack2bf(float lo, float hi) {
    return f2bf(lo) | (f2bf(hi) << 16);
}

// ---------------------------------------------------------------------------
// Pack Wx / Wh (f32 [512,512], row = output n, col = k) into WMMA B-fragment
// order, bf16. dword index d = ((ntile*16 + kchunk)*256) + lane*8 + p, where
// (VGPR p, lane L) = {W[n0+L%16][32c+k], W[n0+L%16][32c+k+1]}, k = 8*(L/16)+2p+(p>=4?8:0) -> each lane's 8
// dwords are contiguous, so the hot loops do two b128 loads per fragment.
// ---------------------------------------------------------------------------
__global__ void prep_pack_w_kernel(const float* __restrict__ Wx,
                                   const float* __restrict__ Wh,
                                   uint32_t* __restrict__ WxB,
                                   uint32_t* __restrict__ WhB) {
    int d = blockIdx.x * blockDim.x + threadIdx.x;     // 0 .. 262143
    const float* W; uint32_t* dst;
    if (d < 131072) { W = Wx; dst = WxB; }
    else            { W = Wh; dst = WhB; d -= 131072; }
    int w   = d >> 12;        // N-tile (0..31)
    int rem = d & 4095;
    int c   = rem >> 8;       // K chunk (0..15)
    int r   = rem & 255;
    int L   = r >> 3;         // lane  (0..31)
    int p   = r & 7;          // VGPR  (0..7)  == K pair
    int n   = w * 16 + (L & 15);
    int k   = 32 * c + ((L < 16) ? 0 : 8) + 2 * p + ((p >= 4) ? 8 : 0);
    dst[d] = pack2bf(W[n * HD + k], W[n * HD + k + 1]);
}

// ---------------------------------------------------------------------------
// Pack x (f32 [32768,512]) into WMMA A-fragment order, bf16.
// dword d = ((mtile*16 + kchunk)*256) + lane*8 + p; A element e of lane L:
// row = mtile*16 + (L%16), K = 32c + off1(L) + e + (e>=8 ? 8 : 0),
// off1 = (L<16 ? 0 : 8). Pairs (2p,2p+1) never straddle the e=8 boundary.
// ---------------------------------------------------------------------------
__global__ void prep_pack_x_kernel(const float* __restrict__ x,
                                   uint32_t* __restrict__ xA) {
    int d = blockIdx.x * blockDim.x + threadIdx.x;     // 0 .. 8388607
    int tm  = d >> 12;        // M-tile (0..2047)
    int rem = d & 4095;
    int c   = rem >> 8;       // K chunk (0..15)
    int r   = rem & 255;
    int L   = r >> 3;         // lane (0..31)
    int p   = r & 7;          // VGPR (0..7)
    int off1 = (L < 16) ? 0 : 8;
    int e0   = 2 * p;
    int row  = tm * 16 + (L & 15);
    int k    = 32 * c + off1 + e0 + ((e0 >= 8) ? 8 : 0);
    const float* src = x + (size_t)row * HD + k;
    xA[d] = pack2bf(src[0], src[1]);
}

// ---------------------------------------------------------------------------
// xp = x @ Wx.T + b   via bf16 WMMA, f32 accumulate.
// Each wave: one 16-row M-strip x four 16-col N-tiles (A reused 4x, four
// independent accumulator chains for XDL ILP).
// 2048 M-strips x 8 N-groups = 16384 waves = 4096 blocks x 128 threads.
// ---------------------------------------------------------------------------
__global__ void xp_gemm_kernel(const uint32_t* __restrict__ xA,
                               const uint32_t* __restrict__ WxB,
                               const float* __restrict__ bias,
                               float* __restrict__ xp) {
    int gwave = (blockIdx.x * blockDim.x + threadIdx.x) >> 5;
    int lane  = threadIdx.x & 31;
    int tm    = gwave >> 3;                  // 0..2047
    int tn4   = (gwave & 7) * 4;             // base N-tile (0,4,..,28)

    f32x8 acc[4] = {{}, {}, {}, {}};
    #pragma unroll
    for (int c = 0; c < 16; ++c) {
        const uint4* Ap = (const uint4*)(xA + (size_t)(tm * 16 + c) * 256 + lane * 8);
        Frag16 a; a.q[0] = Ap[0]; a.q[1] = Ap[1];
        #pragma unroll
        for (int j = 0; j < 4; ++j) {
            const uint4* Bp = (const uint4*)(WxB + (size_t)((tn4 + j) * 16 + c) * 256 + lane * 8);
            Frag16 b; b.q[0] = Bp[0]; b.q[1] = Bp[1];
            acc[j] = __builtin_amdgcn_wmma_f32_16x16x32_bf16(
                         false, a.v, false, b.v, (short)0, acc[j], false, false);
        }
    }
    int rb = tm * 16 + ((lane < 16) ? 0 : 8);
    #pragma unroll
    for (int j = 0; j < 4; ++j) {
        int   col = (tn4 + j) * 16 + (lane & 15);
        float bv  = bias[col];
        CFrag cf; cf.v = acc[j];
        #pragma unroll
        for (int r = 0; r < 8; ++r)
            xp[(size_t)(rb + r) * HD + col] = cf.f[r] + bv;
    }
}

// ---------------------------------------------------------------------------
// Persistent bidirectional scan. One workgroup, 32 waves (1024 threads).
// Wave w owns outputs n0 = w*16 for BOTH directions: A rows with even M carry
// h_fwd, odd M carry h_rev (replicated), so C row 0 = y_fwd, row 1 = y_rev.
// Per step: 16 bf16 WMMAs over K=512 split into TWO independent accumulator
// chains (even/odd chunks) to cover XDL latency; Wh streamed from L2 in
// packed B-fragment order; h double-buffered in LDS as packed bf16.
// ---------------------------------------------------------------------------
__global__ void __launch_bounds__(1024)
scan_kernel(const float* __restrict__ xp, const uint32_t* __restrict__ WhB,
            float* __restrict__ hidden, float* __restrict__ hidden_rev) {
    __shared__ __align__(16) uint16_t hls[2][2][HD];   // [buf][dir][n], bf16 bits
    int tid  = threadIdx.x;
    int w    = tid >> 5;
    int lane = tid & 31;
    int nl   = lane & 15;
    int n    = w * 16 + nl;
    int dir  = nl & 1;                     // row parity: even M = fwd, odd = rev
    int aoff = (lane < 16) ? 0 : 1;        // uint4 sub-offset per A layout

    for (int i = tid; i < 2 * 2 * HD; i += 1024)
        ((uint16_t*)hls)[i] = 0;           // h0 = 0, both buffers/directions
    __syncthreads();

    for (int t = 0; t < T_SEQ; ++t) {
        int cur = t & 1, nxt = cur ^ 1;
        const uint4* hsrc = (const uint4*)&hls[cur][dir][0];   // 64 uint4s
        f32x8 acc0 = {}, acc1 = {};
        #pragma unroll
        for (int cc = 0; cc < 8; ++cc) {
            // even chunk -> chain 0
            {
                int c  = 2 * cc;
                int qi = 4 * c + aoff;
                Frag16 a; a.q[0] = hsrc[qi]; a.q[1] = hsrc[qi + 2];
                const uint4* Bp = (const uint4*)(WhB + (size_t)(w * 16 + c) * 256 + lane * 8);
                Frag16 b; b.q[0] = Bp[0]; b.q[1] = Bp[1];
                acc0 = __builtin_amdgcn_wmma_f32_16x16x32_bf16(
                           false, a.v, false, b.v, (short)0, acc0, false, false);
            }
            // odd chunk -> chain 1
            {
                int c  = 2 * cc + 1;
                int qi = 4 * c + aoff;
                Frag16 a; a.q[0] = hsrc[qi]; a.q[1] = hsrc[qi + 2];
                const uint4* Bp = (const uint4*)(WhB + (size_t)(w * 16 + c) * 256 + lane * 8);
                Frag16 b; b.q[0] = Bp[0]; b.q[1] = Bp[1];
                acc1 = __builtin_amdgcn_wmma_f32_16x16x32_bf16(
                           false, a.v, false, b.v, (short)0, acc1, false, false);
            }
        }
        CFrag c0, c1; c0.v = acc0; c1.v = acc1;
        float yf = c0.f[0] + c1.f[0];      // C row 0 -> y_fwd[n]
        float yr = c0.f[1] + c1.f[1];      // C row 1 -> y_rev[n]
        if (lane < 16) {
            float hf = tanhf(xp[(size_t)t * HD + n] + yf);
            float hr = tanhf(xp[(size_t)(T_SEQ - 1 - t) * HD + n] + yr);
            hidden    [(size_t)t * HD + n] = hf;
            hidden_rev[(size_t)t * HD + n] = hr;
            hls[nxt][0][n] = (uint16_t)f2bf(hf);
            hls[nxt][1][n] = (uint16_t)f2bf(hr);
        }
        __syncthreads();
    }
}

// ---------------------------------------------------------------------------
// y = Wout @ concat(h_fwd_final, h_rev_final) + bout   (512 x 1024 GEMV)
// ---------------------------------------------------------------------------
__global__ void yout_kernel(const float* __restrict__ Wout,
                            const float* __restrict__ bout,
                            const float* __restrict__ hidden,
                            const float* __restrict__ hidden_rev,
                            float* __restrict__ y) {
    __shared__ float hcat[2 * HD];
    int tid = threadIdx.x;                 // 512 threads
    hcat[tid]      = hidden    [(size_t)(T_SEQ - 1) * HD + tid];
    hcat[HD + tid] = hidden_rev[(size_t)(T_SEQ - 1) * HD + tid];
    __syncthreads();
    float acc = bout[tid];
    #pragma unroll 8
    for (int k = 0; k < 2 * HD; ++k)
        acc += Wout[(size_t)tid * (2 * HD) + k] * hcat[k];
    y[tid] = acc;
}

extern "C" void kernel_launch(void* const* d_in, const int* in_sizes, int n_in,
                              void* d_out, int out_size, void* d_ws, size_t ws_size,
                              hipStream_t stream) {
    const float* x    = (const float*)d_in[0];
    const float* Wx   = (const float*)d_in[1];
    const float* Wh   = (const float*)d_in[2];
    const float* b    = (const float*)d_in[3];
    const float* Wout = (const float*)d_in[4];
    const float* bout = (const float*)d_in[5];

    float* out        = (float*)d_out;
    float* y          = out;                                   // [512]
    float* hidden     = out + 512;                             // [32768*512]
    float* hidden_rev = hidden + (size_t)T_SEQ * HD;           // [32768*512]

    char*     ws  = (char*)d_ws;
    float*    xp  = (float*)ws;                                // 64 MB
    uint32_t* WxB = (uint32_t*)(ws + (size_t)T_SEQ * HD * sizeof(float));
    uint32_t* WhB = WxB + 131072;                              // 512 KB each
    uint32_t* xA  = WhB + 131072;                              // 32 MB

    prep_pack_w_kernel<<<1024, 256, 0, stream>>>(Wx, Wh, WxB, WhB);
    prep_pack_x_kernel<<<32768, 256, 0, stream>>>(x, xA);
    xp_gemm_kernel    <<<4096, 128, 0, stream>>>(xA, WxB, b, xp);
    scan_kernel       <<<1, 1024, 0, stream>>>(xp, WhB, hidden, hidden_rev);
    yout_kernel       <<<1, 512, 0, stream>>>(Wout, bout, hidden, hidden_rev, y);
}
